// GaussianRenderer_17592186044888
// MI455X (gfx1250) — matrix operand order repaired and checked by reading, not compiled, not run on hardware
//
#include <hip/hip_runtime.h>
#include <hip/hip_bf16.h>
#include <math.h>

#define Bn 8
#define Nn 64
#define Tt 1024
#define Kk 12
#define SHn 9
#define HIDn 32

typedef __attribute__((ext_vector_type(16))) _Float16 v16h;
typedef __attribute__((ext_vector_type(8)))  float    v8f;
typedef __attribute__((ext_vector_type(4)))  unsigned int v4u;
typedef __attribute__((ext_vector_type(8)))  int      v8i;
typedef __attribute__((ext_vector_type(4)))  int      v4i;

__device__ __forceinline__ float softplusf(float x) {
  return fmaxf(x, 0.f) + log1pf(__expf(-fabsf(x)));
}
__device__ __forceinline__ float gelu_exact(float x) {
  return 0.5f * x * (1.f + erff(x * 0.7071067811865476f));
}
__device__ __forceinline__ void norm3(float& x, float& y, float& z, float eps) {
  float n = sqrtf(x * x + y * y + z * z);
  float inv = 1.f / fmaxf(n, eps);
  x *= inv; y *= inv; z *= inv;
}

// compile-time im2col offset into a zero-padded [C][10][10] LDS image:
// K index g = ic*9 + ky*3 + kx  ->  ic*100 + ky*10 + kx   (add base = py*10+px)
__device__ constexpr int ctab(int g) {
  return (g / 9) * 100 + ((g % 9) / 3) * 10 + ((g % 9) % 3);
}

// ---------------------------------------------------------------------------
// Kernel 0: convert conv weights to f16 once, in WMMA A-operand [oc][K] layout.
// wh[k] = { aw1: 32 x 32 (K=18 zero-padded to 32), aw2: 32 x 288 }  (10240 halves)
// ---------------------------------------------------------------------------
__global__ void wcvt_kernel(const float* __restrict__ w1, const float* __restrict__ w2,
                            _Float16* __restrict__ wh) {
  int idx = blockIdx.x * blockDim.x + threadIdx.x;
  if (idx >= Kk * 10240) return;
  int k = idx / 10240;
  int i = idx - k * 10240;
  _Float16 v;
  if (i < 1024) {
    int oc = i >> 5, kk = i & 31;
    v = (kk < 18) ? (_Float16)w1[(size_t)(k * 32 + oc) * 18 + kk] : (_Float16)0.f;
  } else {
    v = (_Float16)w2[(size_t)k * 9216 + (i - 1024)];
  }
  wh[idx] = v;
}

// ---------------------------------------------------------------------------
// Kernel 1: per (b,k,t) lead vector, camera-plane basis e1/e2, SH basis Y[9]
// lead_ws[(b*K+k)*T+t][18] = {e1[3], e2[3], lv[3], Y[9]}
// ---------------------------------------------------------------------------
__global__ void lead_kernel(const float* __restrict__ chest_offset,
                            const float* __restrict__ resp_freq,
                            const float* __restrict__ resp_vec,
                            const float* __restrict__ limb_ra,
                            const float* __restrict__ limb_la,
                            const float* __restrict__ limb_ll,
                            const float* __restrict__ chest_vectors,
                            float* __restrict__ lead_ws) {
  int idx = blockIdx.x * blockDim.x + threadIdx.x;
  if (idx >= Bn * Kk * Tt) return;
  int b = idx / (Kk * Tt);
  int rem = idx - b * (Kk * Tt);
  int k = rem / Tt;
  int t = rem - k * Tt;
  float tv = (float)t * (1.f / (Tt - 1));

  float bx, by, bz;
  if (k < 6) {
    float rax = limb_ra[0], ray = limb_ra[1], raz = limb_ra[2];
    float lax = limb_la[0], lay = limb_la[1], laz = limb_la[2];
    float llx = limb_ll[0], lly = limb_ll[1], llz = limb_ll[2];
    switch (k) {
      case 0: bx = lax - rax; by = lay - ray; bz = laz - raz; break;
      case 1: bx = llx - rax; by = lly - ray; bz = llz - raz; break;
      case 2: bx = llx - lax; by = lly - lay; bz = llz - laz; break;
      case 3: bx = rax - 0.5f * (lax + llx); by = ray - 0.5f * (lay + lly); bz = raz - 0.5f * (laz + llz); break;
      case 4: bx = lax - 0.5f * (rax + llx); by = lay - 0.5f * (ray + lly); bz = laz - 0.5f * (raz + llz); break;
      default: bx = llx - 0.5f * (rax + lax); by = lly - 0.5f * (ray + lay); bz = llz - 0.5f * (raz + laz); break;
    }
    norm3(bx, by, bz, 1e-6f);
  } else {
    int c = k - 6;
    float cx = chest_vectors[c * 3 + 0], cy = chest_vectors[c * 3 + 1], cz = chest_vectors[c * 3 + 2];
    norm3(cx, cy, cz, 1e-12f);
    bx = cx + chest_offset[(b * 6 + c) * 3 + 0];
    by = cy + chest_offset[(b * 6 + c) * 3 + 1];
    bz = cz + chest_offset[(b * 6 + c) * 3 + 2];
    norm3(bx, by, bz, 1e-12f);
  }

  float sw = sinf(6.283185307179586f * resp_freq[b] * tv);
  float lx = bx + resp_vec[b * 3 + 0] * sw;
  float ly = by + resp_vec[b * 3 + 1] * sw;
  float lz = bz + resp_vec[b * 3 + 2] * sw;
  norm3(lx, ly, lz, 1e-12f);

  float e1x = 0.f, e1y = lz, e1z = -ly;           // L x (1,0,0)
  float n1 = sqrtf(e1y * e1y + e1z * e1z);
  if (n1 < 1e-4f) { e1x = -lz; e1y = 0.f; e1z = lx; }  // L x (0,1,0)
  norm3(e1x, e1y, e1z, 1e-6f);
  float e2x = e1y * lz - e1z * ly;
  float e2y = e1z * lx - e1x * lz;
  float e2z = e1x * ly - e1y * lx;
  norm3(e2x, e2y, e2z, 1e-6f);

  float* o = lead_ws + (size_t)idx * 18;
  o[0] = e1x; o[1] = e1y; o[2] = e1z;
  o[3] = e2x; o[4] = e2y; o[5] = e2z;
  o[6] = lx;  o[7] = ly;  o[8] = lz;
  o[9]  = 0.282095f;
  o[10] = -0.488603f * ly;
  o[11] =  0.488603f * lz;
  o[12] = -0.488603f * lx;
  o[13] =  1.092548f * lx * ly;
  o[14] = -1.092548f * ly * lz;
  o[15] =  0.315392f * (3.f * lz * lz - 1.f);
  o[16] = -1.092548f * lx * lz;
  o[17] =  0.546274f * (lx * lx - ly * ly);
}

// ---------------------------------------------------------------------------
// Kernel 2: per (b,n,t) gaussian envelope, p_hat, sh_dyn
// pernt_ws[(b*T+t)*N+n][13] = {gauss, p_hat[3], sh_dyn[9]}
// ---------------------------------------------------------------------------
__global__ void point_kernel(const float* __restrict__ mu, const float* __restrict__ sigma0,
                             const float* __restrict__ sigma_vel, const float* __restrict__ amplitude,
                             const float* __restrict__ sh_base, const float* __restrict__ sh_vel,
                             const float* __restrict__ p0, const float* __restrict__ p_vel,
                             float* __restrict__ pernt_ws) {
  int idx = blockIdx.x * blockDim.x + threadIdx.x;
  if (idx >= Bn * Tt * Nn) return;
  int b = idx / (Tt * Nn);
  int r = idx - b * (Tt * Nn);
  int t = r / Nn;
  int n = r - t * Nn;
  int bn = b * Nn + n;
  float tv = (float)t * (1.f / (Tt - 1));
  float dt = tv - mu[bn];
  float st = softplusf(sigma0[bn] + sigma_vel[bn] * dt) + 0.001f;
  float z = dt / st;
  float g = amplitude[bn] * __expf(-0.5f * z * z);
  float px = p0[bn * 3 + 0] + p_vel[bn * 3 + 0] * dt;
  float py = p0[bn * 3 + 1] + p_vel[bn * 3 + 1] * dt;
  float pz = p0[bn * 3 + 2] + p_vel[bn * 3 + 2] * dt;
  norm3(px, py, pz, 1e-6f);
  float* o = pernt_ws + (size_t)idx * 13;
  o[0] = g; o[1] = px; o[2] = py; o[3] = pz;
#pragma unroll
  for (int m = 0; m < 9; ++m) o[4 + m] = sh_base[bn * 9 + m] + sh_vel[bn * 9 + m] * dt;
}

// ---------------------------------------------------------------------------
// Kernel 3 (fused): one 64-thread WG (2 waves) per image (b,k,t).
// TDM stages 20KB of f16 weights into LDS while the wave computes the splat;
// conv1/conv2 are WMMA implicit-GEMMs over zero-padded LDS images (branchless
// im2col with compile-time offset tables).
// ---------------------------------------------------------------------------
__global__ __launch_bounds__(64) void splat_cnn_kernel(
    const float* __restrict__ pernt_ws, const float* __restrict__ lead_ws,
    const _Float16* __restrict__ wh,
    const float* __restrict__ b1, const float* __restrict__ b2,
    const float* __restrict__ wf, const float* __restrict__ bf,
    const float* __restrict__ tau_raw, float* __restrict__ Vout) {
  __shared__ float ldE[18];
  __shared__ float4 pk[64];                         // {u, v, a, a*sh} per splat
  __shared__ float imgp[2 * 100 + 80];              // padded [2][10][10] + zero pad region
  __shared__ __align__(16) _Float16 awAll[10240];   // aw1 [32][32] then aw2 [32][288]
  __shared__ __align__(16) _Float16 hid1p[32 * 100];// padded [32][10][10]
  __shared__ float b1S[32], b2S[32], wfS[32];
  __shared__ float red;

  int blk = blockIdx.x;                             // (b*K+k)*T + t
  int b = blk / (Kk * Tt);
  int rem = blk - b * (Kk * Tt);
  int k = rem / Tt;
  int t = rem - k * Tt;
  int tid = threadIdx.x;
  int lane = tid & 31;
  int wv = tid >> 5;

  // ---- TDM: DMA this lead's pre-converted f16 weights (20480 B) into LDS ----
  if (wv == 0) {
    const _Float16* src = wh + (size_t)k * 10240;
    unsigned long long ga = (unsigned long long)(uintptr_t)src;
    unsigned int lds_off = (unsigned int)(uintptr_t)(&awAll[0]); // LDS aperture: low 32 = offset
    const unsigned int LEN8 = 20480u / 8u;                       // 2560 x 8B units
    v4u g0;
    g0[0] = 1u;                                       // count=1 valid descriptor
    g0[1] = lds_off;                                  // lds_addr
    g0[2] = (unsigned int)ga;                         // global_addr[31:0]
    g0[3] = ((unsigned int)(ga >> 32) & 0x01FFFFFFu)  // global_addr[56:32]
            | 0x80000000u;                            // type=2 (image)
    v8i g1;
    g1[0] = 3 << 16;                                  // workgroup_mask=0, data_size=8B
    g1[1] = (int)((LEN8 & 0xFFFFu) << 16);            // tensor_dim0[15:0]
    g1[2] = (int)((LEN8 >> 16) | (1u << 16));         // tensor_dim0[31:16], tensor_dim1=1
    g1[3] = (int)(LEN8 << 16);                        // tile_dim0 = 2560
    g1[4] = 1;                                        // tile_dim1 = 1
    g1[5] = (int)LEN8;                                // tensor_dim0_stride
    g1[6] = 0;
    g1[7] = 0;
    v4i z4; z4[0] = 0; z4[1] = 0; z4[2] = 0; z4[3] = 0;
    v8i z8; z8[0] = 0; z8[1] = 0; z8[2] = 0; z8[3] = 0;
    z8[4] = 0; z8[5] = 0; z8[6] = 0; z8[7] = 0;
    __builtin_amdgcn_tensor_load_to_lds(g0, g1, z4, z4, z8, 0);  // 6-arg (clang-23) form
  }

  // ---- stage small data, zero padded images ----
  if (tid < 18) ldE[tid] = lead_ws[(size_t)blk * 18 + tid];
  if (tid < 32) {
    b1S[tid] = b1[k * 32 + tid];
    b2S[tid] = b2[k * 32 + tid];
    wfS[tid] = wf[k * 32 + tid];
  }
  if (tid == 0) red = 0.f;
  for (int i = tid; i < 2 * 100 + 80; i += 64) imgp[i] = 0.f;
  {
    unsigned int* hz = (unsigned int*)hid1p;          // 3200 halves = 1600 dwords
    for (int i = tid; i < 1600; i += 64) hz[i] = 0u;
  }
  __syncthreads();

  // ---- per-splat values: thread tid handles splat n = tid ----
  {
    const float* pc = pernt_ws + ((size_t)(b * Tt + t) * Nn + tid) * 13;
    float g = pc[0];
    float hx = pc[1], hy = pc[2], hz = pc[3];
    float uu = tanhf(hx * ldE[0] + hy * ldE[1] + hz * ldE[2]);
    float vv = tanhf(hx * ldE[3] + hy * ldE[4] + hz * ldE[5]);
    float cp = hx * ldE[6] + hy * ldE[7] + hz * ldE[8];
    float sh = 0.f;
#pragma unroll
    for (int m = 0; m < 9; ++m) sh += pc[4 + m] * ldE[9 + m];
    float aa = g * fmaxf(cp, 0.f);                    // gauss * hem
    pk[tid] = make_float4(uu, vv, aa, aa * sh);
  }
  __syncthreads();

  // ---- splat: thread tid owns grid cell (py,px); write padded interior ----
  int py = tid >> 3, px = tid & 7;
  {
    float tau = softplusf(tau_raw[0]) + 0.06f;
    float invq = 0.5f / (tau * tau);
    float gxv = -1.f + (float)px * (2.f / 7.f);
    float gyv = -1.f + (float)py * (2.f / 7.f);
    float sp = 0.f, ss = 0.f;
#pragma unroll 4
    for (int n = 0; n < Nn; ++n) {
      float4 q = pk[n];
      float du = q.x - gxv, dv = q.y - gyv;
      float kq = __expf(-(du * du + dv * dv) * invq);
      sp += q.z * kq;
      ss += q.w * kq;
    }
    int pi = (py + 1) * 10 + (px + 1);
    imgp[pi] = sp;
    imgp[100 + pi] = ss;
  }
  // drain the weight DMA before WMMA consumes awAll
  if (wv == 0) __builtin_amdgcn_s_wait_tensorcnt(0);
  __syncthreads();

  // WMMA wave32 fragment coordinates (ISA 7.12.2):
  //  A: lane -> row M=lane&15; K = (lane>>4)*16 + e (contiguous order used in this code)
  //  B: lane -> col N=lane&15; K = (lane>>4)*16 + e (this code's order, same as A)
  //  C/D: lane -> col N=lane&15; VGPR j -> row M=(lane>>4)*8+j
  int col = lane & 15;
  int hf = lane >> 4;
  const _Float16* aw1 = awAll;
  const _Float16* aw2 = awAll + 1024;

  // ---- conv1: 2ch -> 32ch 3x3 SAME; K=18 padded to 32 (one WMMA per tile) ----
#pragma unroll
  for (int nti = 0; nti < 2; ++nti) {
    int nt = 2 * wv + nti;
    int p = nt * 16 + col;
    int base = (p >> 3) * 10 + (p & 7);               // unpadded (py,px) -> padded base
    v16h bfr;
#pragma unroll
    for (int e = 0; e < 16; ++e) {
      constexpr int ZOFF = 200;                       // zero-region start in imgp
      int off0 = ctab(e);                             // kk = e (<18)
      int off1 = (16 + e < 18) ? ctab(16 + e) : ZOFF; // kk = 16+e, pad K -> zeros
      bfr[e] = (_Float16)imgp[(hf ? off1 : off0) + base];
    }
#pragma unroll
    for (int mt = 0; mt < 2; ++mt) {
      v16h afr;
      int oc0 = mt * 16 + col;
#pragma unroll
      for (int e = 0; e < 16; ++e) {
        int Kq = hf * 16 + e;
        afr[e] = aw1[oc0 * 32 + Kq];
      }
      v8f acc = {};
      acc = __builtin_amdgcn_wmma_f32_16x16x32_f16(false, afr, false, bfr,
                                                   (short)0, acc, false, false);
#pragma unroll
      for (int j = 0; j < 8; ++j) {
        int oc = mt * 16 + hf * 8 + j;
        float x = acc[j] + b1S[oc];
        hid1p[oc * 100 + base + 11] = (_Float16)gelu_exact(x);  // interior (+1,+1)
      }
    }
  }
  __syncthreads();

  // ---- conv2: 32ch -> 32ch 3x3 SAME; K=288 = 9 WMMA K-steps, B reused over M ----
  float partial = 0.f;
#pragma unroll
  for (int nti = 0; nti < 2; ++nti) {
    int nt = 2 * wv + nti;
    int p = nt * 16 + col;
    int base = (p >> 3) * 10 + (p & 7);
    v8f acc0 = {};
    v8f acc1 = {};
#pragma unroll
    for (int s = 0; s < 9; ++s) {
      v16h bfr;
#pragma unroll
      for (int e = 0; e < 16; ++e) {
        int off0 = ctab(s * 32 + e);                  // constants folded at compile time
        int off1 = ctab(s * 32 + 16 + e);
        bfr[e] = hid1p[(hf ? off1 : off0) + base];
      }
      v16h afr0, afr1;
#pragma unroll
      for (int e = 0; e < 16; ++e) {
        int Kq = s * 32 + hf * 16 + e;
        afr0[e] = aw2[col * 288 + Kq];
        afr1[e] = aw2[(16 + col) * 288 + Kq];
      }
      acc0 = __builtin_amdgcn_wmma_f32_16x16x32_f16(false, afr0, false, bfr,
                                                    (short)0, acc0, false, false);
      acc1 = __builtin_amdgcn_wmma_f32_16x16x32_f16(false, afr1, false, bfr,
                                                    (short)0, acc1, false, false);
    }
#pragma unroll
    for (int j = 0; j < 8; ++j) {
      int oca = hf * 8 + j;
      partial += wfS[oca] * gelu_exact(acc0[j] + b2S[oca]);
      int ocb = 16 + hf * 8 + j;
      partial += wfS[ocb] * gelu_exact(acc1[j] + b2S[ocb]);
    }
  }
  atomicAdd(&red, partial);                           // ds_add_f32
  __syncthreads();
  if (tid == 0) Vout[blk] = red * (1.f / 64.f) + bf[k];
}

// ---------------------------------------------------------------------------
extern "C" void kernel_launch(void* const* d_in, const int* in_sizes, int n_in,
                              void* d_out, int out_size, void* d_ws, size_t ws_size,
                              hipStream_t stream) {
  const float* mu            = (const float*)d_in[0];
  const float* sigma0        = (const float*)d_in[1];
  const float* sigma_vel     = (const float*)d_in[2];
  const float* amplitude     = (const float*)d_in[3];
  const float* sh_base       = (const float*)d_in[4];
  const float* sh_vel        = (const float*)d_in[5];
  const float* p0            = (const float*)d_in[6];
  const float* p_vel         = (const float*)d_in[7];
  const float* chest_offset  = (const float*)d_in[8];
  const float* resp_freq     = (const float*)d_in[9];
  const float* resp_vec      = (const float*)d_in[10];
  const float* splat_tau_raw = (const float*)d_in[11];
  const float* limb_ra       = (const float*)d_in[12];
  const float* limb_la       = (const float*)d_in[13];
  const float* limb_ll       = (const float*)d_in[14];
  const float* chest_vectors = (const float*)d_in[15];
  const float* w1            = (const float*)d_in[16];
  const float* b1            = (const float*)d_in[17];
  const float* w2            = (const float*)d_in[18];
  const float* b2            = (const float*)d_in[19];
  const float* wf            = (const float*)d_in[20];
  const float* bf            = (const float*)d_in[21];
  (void)in_sizes; (void)n_in; (void)out_size; (void)ws_size;

  float* lead_ws  = (float*)d_ws;                           // 98304*18 f32 (16B aligned)
  float* pernt_ws = lead_ws + (size_t)Bn * Kk * Tt * 18;    // 524288*13 f32
  _Float16* wh_ws = (_Float16*)(pernt_ws + (size_t)Bn * Tt * Nn * 13); // 122880 f16

  int n0 = Kk * 10240;     // 122880
  int n1 = Bn * Kk * Tt;   // 98304
  int n2 = Bn * Tt * Nn;   // 524288
  wcvt_kernel<<<(n0 + 255) / 256, 256, 0, stream>>>(w1, w2, wh_ws);
  lead_kernel<<<(n1 + 255) / 256, 256, 0, stream>>>(chest_offset, resp_freq, resp_vec,
                                                    limb_ra, limb_la, limb_ll,
                                                    chest_vectors, lead_ws);
  point_kernel<<<(n2 + 255) / 256, 256, 0, stream>>>(mu, sigma0, sigma_vel, amplitude,
                                                     sh_base, sh_vel, p0, p_vel, pernt_ws);
  splat_cnn_kernel<<<n1, 64, 0, stream>>>(pernt_ws, lead_ws, wh_ws, b1, b2, wf, bf,
                                          splat_tau_raw, (float*)d_out);
}
